// crossAttention_47708496724255
// MI455X (gfx1250) — compile-verified, matrix operand order checked by reading, not run on hardware
//
#include <hip/hip_runtime.h>

// MI455X / gfx1250, wave32. All GEMM-like math via v_wmma_f32_16x16x32_bf16.

typedef __attribute__((ext_vector_type(16))) __bf16 v16bf;
typedef __attribute__((ext_vector_type(8)))  float  v8f;

#define GN 102400   // H*W
#define HH 320
#define WW 320
#define CC 192

union Frag16 { uint4 u[2]; v16bf v; };
union U4B    { uint4 u; __bf16 h[8]; };

// ---------------------------------------------------------------------------
// prep: weights fp32->bf16, zero norm/S accumulators (every call, stateless)
// ---------------------------------------------------------------------------
__global__ __launch_bounds__(256) void prep_kernel(
    const float* __restrict__ kv_w, const float* __restrict__ q_w,
    __bf16* __restrict__ wkv, __bf16* __restrict__ wq, float* __restrict__ z)
{
    int i = blockIdx.x * 256 + threadIdx.x;
    if (i < 384 * 192) wkv[i] = (__bf16)kv_w[i];
    if (i < 192 * 192) wq[i]  = (__bf16)q_w[i];
    if (i < 9600)      z[i]   = 0.0f;   // normk(192) + normq(192) + S(9216)
}

// ---------------------------------------------------------------------------
// pe_add: exact port of the raw [1,W,H,C]->[1,C,H,W] reshape of the PE.
// flat idx = c*H*W + h*W + w re-read as (w', h', c') of the [W,H,C] tensor.
// ---------------------------------------------------------------------------
__global__ __launch_bounds__(256) void pe_add_kernel(
    const float* __restrict__ x, const float* __restrict__ y,
    __bf16* __restrict__ xb, __bf16* __restrict__ yb)
{
    int idx = blockIdx.x * 256 + threadIdx.x;           // 0 .. 192*102400-1
    int w1 = idx / (HH * CC);                           // 61440
    int r  = idx - w1 * (HH * CC);
    int h1 = r / CC;
    int c1 = r - h1 * CC;
    int base = (c1 < 96) ? c1 : c1 - 96;
    int cc   = (base < 48) ? base : base - 48;
    float pos = (c1 < 96) ? (float)h1 : (float)w1;      // pe_h uses y, pe_w uses x
    int j2 = cc >> 1;
    float dv = __expf(-0.19188209108f * (float)(2 * j2)); // exp(-ln(1e4)/48 * 2j)
    float a  = pos * dv;
    float pe = (cc & 1) ? __cosf(a) : __sinf(a);
    xb[idx] = (__bf16)(x[idx] + pe);
    yb[idx] = (__bf16)(y[idx] + pe);
}

// ---------------------------------------------------------------------------
// Tiled WMMA GEMM: Out[M][GN] = A[M][192] * B[192][GN] + bias
// block: 256 thr (8 waves), tile 64(M) x 128(N); per wave 16x64 (4 accs).
// B tile is stored transposed in LDS so both fragments are 2x b128 LDS loads
// per lane (16-bit A layout: K offsets hi*8+{0..7} and hi*8+16+{0..7}).
// ---------------------------------------------------------------------------
template <bool OUTF32>
__global__ __launch_bounds__(256) void gemm_kernel(
    const __bf16* __restrict__ A,    // [M][192]
    const __bf16* __restrict__ B,    // [192][GN]
    const float*  __restrict__ bias, // [M]
    void* __restrict__ Out)          // [M][GN] bf16 or f32
{
    __shared__ __bf16 Alds[64][40];   // 64 x 32 tile, padded (80B rows, 16B aligned)
    __shared__ __bf16 Blds[128][40];  // transposed 32 x 128 tile -> [n][k]

    const int t    = threadIdx.x;
    const int n0   = blockIdx.x * 128;
    const int m0   = blockIdx.y * 64;
    const int wave = t >> 5, lane = t & 31;
    const int mw   = (wave & 3) * 16;
    const int nwB  = (wave >> 2) * 64;
    const int m    = lane & 15;
    const int hi   = lane >> 4;

    const int arow = t >> 2;          // 0..63
    const int acg  = (t & 3) * 8;     // 0,8,16,24

    v8f acc[4] = {};

    for (int kk = 0; kk < 192; kk += 32) {
        // A tile: 64x32 bf16, one b128 per thread
        *(uint4*)&Alds[arow][acg] =
            *(const uint4*)(A + (size_t)(m0 + arow) * 192 + kk + acg);
        // B tile: 32x128 bf16, transposed into LDS
        #pragma unroll
        for (int part = 0; part < 2; ++part) {
            int idx = t + part * 256;
            int kr  = idx >> 4;
            int ncg = (idx & 15) * 8;
            U4B u;
            u.u = *(const uint4*)(B + (size_t)(kk + kr) * GN + n0 + ncg);
            #pragma unroll
            for (int j = 0; j < 8; ++j) Blds[ncg + j][kr] = u.h[j];
        }
        __syncthreads();

        Frag16 a;
        a.u[0] = *(const uint4*)&Alds[mw + m][hi * 8];
        a.u[1] = *(const uint4*)&Alds[mw + m][hi * 8 + 16];
        #pragma unroll
        for (int na = 0; na < 4; ++na) {
            Frag16 b;
            const int nn = nwB + na * 16 + m;
            b.u[0] = *(const uint4*)&Blds[nn][hi * 8];
            b.u[1] = *(const uint4*)&Blds[nn][hi * 8 + 16];
            acc[na] = __builtin_amdgcn_wmma_f32_16x16x32_bf16(
                false, a.v, false, b.v, (short)0, acc[na], false, false);
        }
        __syncthreads();
    }

    #pragma unroll
    for (int na = 0; na < 4; ++na) {
        #pragma unroll
        for (int r2 = 0; r2 < 8; ++r2) {
            int orow = m0 + mw + r2 + hi * 8;   // D: vgpr r holds rows r, r+8
            int ocol = n0 + nwB + na * 16 + m;
            float val = acc[na][r2] + bias[orow];
            if (OUTF32) ((float*) Out)[(size_t)orow * GN + ocol] = val;
            else        ((__bf16*)Out)[(size_t)orow * GN + ocol] = (__bf16)val;
        }
    }
}

// ---------------------------------------------------------------------------
// depthwise 3x3 (zero pad), fp32 accumulate; splits kv into k/v, produces q;
// block-reduces sum of squares per channel for the L2 norms.
// grid: (400, 576) ; block 256 (one channel per block.y, 256 pixels)
// ---------------------------------------------------------------------------
__global__ __launch_bounds__(256) void dwconv_kernel(
    const __bf16* __restrict__ kvpre, const __bf16* __restrict__ qpre,
    const float* __restrict__ kv_dw_w, const float* __restrict__ kv_dw_b,
    const float* __restrict__ q_dw_w,  const float* __restrict__ q_dw_b,
    __bf16* __restrict__ kbuf, __bf16* __restrict__ vbuf, __bf16* __restrict__ qbuf,
    float* __restrict__ normk, float* __restrict__ normq)
{
    __shared__ float red[256];
    const int ch  = blockIdx.y;                     // 0..575
    const int pix = blockIdx.x * 256 + threadIdx.x;
    const int h   = pix / WW, w = pix - h * WW;

    const __bf16* src; const float* wt; float bsv; __bf16* dst;
    if (ch < 2 * CC) {
        src = kvpre + (size_t)ch * GN; wt = kv_dw_w + ch * 9; bsv = kv_dw_b[ch];
        dst = (ch < CC) ? (kbuf + (size_t)ch * GN) : (vbuf + (size_t)(ch - CC) * GN);
    } else {
        int c2 = ch - 2 * CC;
        src = qpre + (size_t)c2 * GN; wt = q_dw_w + c2 * 9; bsv = q_dw_b[c2];
        dst = qbuf + (size_t)c2 * GN;
    }
    float acc = bsv;
    #pragma unroll
    for (int ky = -1; ky <= 1; ++ky) {
        int h2 = h + ky;
        if (h2 < 0 || h2 >= HH) continue;
        #pragma unroll
        for (int kx = -1; kx <= 1; ++kx) {
            int w2 = w + kx;
            if (w2 < 0 || w2 >= WW) continue;
            acc += (float)src[h2 * WW + w2] * wt[(ky + 1) * 3 + (kx + 1)];
        }
    }
    dst[pix] = (__bf16)acc;

    red[threadIdx.x] = acc * acc;
    __syncthreads();
    for (int s2 = 128; s2 > 0; s2 >>= 1) {
        if (threadIdx.x < s2) red[threadIdx.x] += red[threadIdx.x + s2];
        __syncthreads();
    }
    if (threadIdx.x == 0) {
        if (ch < CC)            atomicAdd(&normk[ch], red[0]);
        else if (ch >= 2 * CC)  atomicAdd(&normq[ch - 2 * CC], red[0]);
    }
}

// ---------------------------------------------------------------------------
// Gram: S[h] += q_h(48 x Nchunk) * k_h(48 x Nchunk)^T, split-K with atomics.
// One wave per block keeps all 9 (3x3) 16x16 accumulators of a head.
// Fragments load directly from global (rows are K-contiguous, 2x b128/lane).
// grid: (200, 4), block 32.
// ---------------------------------------------------------------------------
__global__ __launch_bounds__(32) void gram_kernel(
    const __bf16* __restrict__ q, const __bf16* __restrict__ k,
    float* __restrict__ S)
{
    const int head  = blockIdx.y;
    const int nbase = blockIdx.x * 512;
    const int lane  = threadIdx.x;
    const int m     = lane & 15, hi = lane >> 4;

    v8f acc[3][3] = {};
    for (int n0 = nbase; n0 < nbase + 512; n0 += 32) {
        Frag16 a[3], b[3];
        #pragma unroll
        for (int it = 0; it < 3; ++it) {
            const __bf16* qp = q + (size_t)(head * 48 + it * 16 + m) * GN + n0 + hi * 8;
            a[it].u[0] = *(const uint4*)qp;
            a[it].u[1] = *(const uint4*)(qp + 16);
            const __bf16* kp = k + (size_t)(head * 48 + it * 16 + m) * GN + n0 + hi * 8;
            b[it].u[0] = *(const uint4*)kp;
            b[it].u[1] = *(const uint4*)(kp + 16);
        }
        #pragma unroll
        for (int it = 0; it < 3; ++it)
            #pragma unroll
            for (int jt = 0; jt < 3; ++jt)
                acc[it][jt] = __builtin_amdgcn_wmma_f32_16x16x32_bf16(
                    false, a[it].v, false, b[jt].v, (short)0, acc[it][jt], false, false);
    }
    #pragma unroll
    for (int it = 0; it < 3; ++it)
        #pragma unroll
        for (int jt = 0; jt < 3; ++jt)
            #pragma unroll
            for (int r2 = 0; r2 < 8; ++r2) {
                int row = it * 16 + r2 + hi * 8;
                int col = jt * 16 + m;
                atomicAdd(&S[head * 2304 + row * 48 + col], acc[it][jt][r2]);
            }
}

// ---------------------------------------------------------------------------
// attn_proj: rescale S by 1/(||q||*||k||)*tau, softmax rows, then fuse
// W2 = proj_w * blockdiag(attn)  (192x192 bf16). Single workgroup.
// ---------------------------------------------------------------------------
__global__ __launch_bounds__(256) void attn_proj_kernel(
    const float* __restrict__ S, const float* __restrict__ normk,
    const float* __restrict__ normq, const float* __restrict__ temp,
    const float* __restrict__ proj_w, __bf16* __restrict__ w2)
{
    __shared__ float attn[192][48];
    const int r = threadIdx.x;
    if (r < 192) {
        int h = r / 48, c = r - h * 48;
        float nq = fmaxf(sqrtf(normq[r]), 1e-12f);
        float tf = temp[h];
        float row[48];
        float mx = -3.4e38f;
        for (int j = 0; j < 48; ++j) {
            float nk = fmaxf(sqrtf(normk[h * 48 + j]), 1e-12f);
            float v2 = S[h * 2304 + c * 48 + j] / (nq * nk) * tf;
            row[j] = v2;
            mx = fmaxf(mx, v2);
        }
        float s2 = 0.0f;
        for (int j = 0; j < 48; ++j) { row[j] = __expf(row[j] - mx); s2 += row[j]; }
        float inv = 1.0f / s2;
        for (int j = 0; j < 48; ++j) attn[r][j] = row[j] * inv;
    }
    __syncthreads();
    for (int idx = r; idx < 192 * 192; idx += 256) {
        int o = idx / 192, tc = idx - o * 192;
        int h = tc / 48, d = tc - h * 48;
        float s2 = 0.0f;
        #pragma unroll
        for (int c = 0; c < 48; ++c)
            s2 += proj_w[o * 192 + h * 48 + c] * attn[h * 48 + c][d];
        w2[idx] = (__bf16)s2;
    }
}

// ---------------------------------------------------------------------------
extern "C" void kernel_launch(void* const* d_in, const int* in_sizes, int n_in,
                              void* d_out, int out_size, void* d_ws, size_t ws_size,
                              hipStream_t stream)
{
    const float* x        = (const float*)d_in[0];
    const float* y        = (const float*)d_in[1];
    const float* kv_w     = (const float*)d_in[2];
    const float* kv_b     = (const float*)d_in[3];
    const float* kv_dw_w  = (const float*)d_in[4];
    const float* kv_dw_b  = (const float*)d_in[5];
    const float* q_w      = (const float*)d_in[6];
    const float* q_b      = (const float*)d_in[7];
    const float* q_dw_w   = (const float*)d_in[8];
    const float* q_dw_b   = (const float*)d_in[9];
    const float* proj_w   = (const float*)d_in[10];
    const float* proj_b   = (const float*)d_in[11];
    const float* temperature = (const float*)d_in[12];
    (void)in_sizes; (void)n_in; (void)out_size; (void)ws_size;

    // Workspace layout (~237 MB with buffer reuse)
    char* ws = (char*)d_ws;
    size_t off = 0;
    auto alloc = [&](size_t b) { size_t r = off; off += (b + 255) & ~(size_t)255; return r; };
    size_t o_xb  = alloc((size_t)CC * GN * 2);       // bf16 x+pe
    size_t o_yb  = alloc((size_t)CC * GN * 2);       // bf16 y+pe
    size_t o_kvp = alloc((size_t)2 * CC * GN * 2);   // bf16 kv pre-dw
    size_t o_qp  = alloc((size_t)CC * GN * 2);       // bf16 q pre-dw
    size_t o_v   = alloc((size_t)CC * GN * 2);       // bf16 v
    size_t o_wkv = alloc((size_t)384 * 192 * 2);
    size_t o_wq  = alloc((size_t)192 * 192 * 2);
    size_t o_w2  = alloc((size_t)192 * 192 * 2);
    size_t o_z   = alloc((size_t)9600 * 4);          // normk|normq|S

    __bf16* xb    = (__bf16*)(ws + o_xb);
    __bf16* yb    = (__bf16*)(ws + o_yb);
    __bf16* kvpre = (__bf16*)(ws + o_kvp);
    __bf16* qpre  = (__bf16*)(ws + o_qp);
    __bf16* vbuf  = (__bf16*)(ws + o_v);
    __bf16* kbuf  = yb;   // yb dead after q GEMM -> reuse for k
    __bf16* qbuf  = xb;   // xb dead after kv GEMM -> reuse for q
    __bf16* wkv   = (__bf16*)(ws + o_wkv);
    __bf16* wq    = (__bf16*)(ws + o_wq);
    __bf16* w2    = (__bf16*)(ws + o_w2);
    float*  normk = (float*)(ws + o_z);
    float*  normq = normk + 192;
    float*  S     = normq + 192;

    prep_kernel<<<288, 256, 0, stream>>>(kv_w, q_w, wkv, wq, normk);
    pe_add_kernel<<<(CC * GN) / 256, 256, 0, stream>>>(x, y, xb, yb);
    gemm_kernel<false><<<dim3(GN / 128, 6), 256, 0, stream>>>(wkv, xb, kv_b, kvpre);
    gemm_kernel<false><<<dim3(GN / 128, 3), 256, 0, stream>>>(wq,  yb, q_b,  qpre);
    dwconv_kernel<<<dim3(GN / 256, 576), 256, 0, stream>>>(
        kvpre, qpre, kv_dw_w, kv_dw_b, q_dw_w, q_dw_b,
        kbuf, vbuf, qbuf, normk, normq);
    gram_kernel<<<dim3(GN / 512, 4), 32, 0, stream>>>(qbuf, kbuf, S);
    attn_proj_kernel<<<1, 256, 0, stream>>>(S, normk, normq, temperature, proj_w, w2);
    gemm_kernel<true><<<dim3(GN / 128, 3), 256, 0, stream>>>(w2, vbuf, proj_b, (float*)d_out);
}
